// AgentAttention_25383256720171
// MI455X (gfx1250) — matrix operand order repaired and checked by reading, not compiled, not run on hardware
//
#include <hip/hip_runtime.h>
#include <hip/hip_bf16.h>
#include <stdint.h>

#define HEADS 8
#define DH    64
#define DIMK  512
#define MAG   256
#define BATCH 4
#define SEQ   8192

typedef __bf16 bf16_t;
typedef bf16_t          v16bf __attribute__((ext_vector_type(16)));
typedef bf16_t          v2bf  __attribute__((ext_vector_type(2)));
typedef unsigned short  v16us __attribute__((ext_vector_type(16)));
typedef float           v8f   __attribute__((ext_vector_type(8)));
typedef unsigned int    u32x4 __attribute__((ext_vector_type(4)));
typedef int             i32x8 __attribute__((ext_vector_type(8)));
typedef int             i32x4 __attribute__((ext_vector_type(4)));

// ---- fp32 -> bf16 conversion: packed hardware cvt when available ----------
__device__ __forceinline__ unsigned cvt2(float a, float b) {
#if __has_builtin(__builtin_amdgcn_cvt_pk_bf16_f32)
  v2bf r = __builtin_amdgcn_cvt_pk_bf16_f32(a, b);
  return __builtin_bit_cast(unsigned, r);
#else
  union { float f; unsigned u; } ca, cb; ca.f = a; cb.f = b;
  unsigned ua = (ca.u + 0x7FFFu + ((ca.u >> 16) & 1u)) >> 16;
  unsigned ub = (cb.u + 0x7FFFu + ((cb.u >> 16) & 1u)) >> 16;
  return (ua & 0xFFFFu) | (ub << 16);
#endif
}
__device__ __forceinline__ unsigned short f2bf(float f) {
  return (unsigned short)(cvt2(f, f) & 0xFFFFu);
}

__device__ __forceinline__ v8f wmma_bf16(v16us a, v16us b, v8f c) {
  return __builtin_amdgcn_wmma_f32_16x16x32_bf16(
      false, __builtin_bit_cast(v16bf, a),
      false, __builtin_bit_cast(v16bf, b),
      (short)0, c, false, false);
}

__device__ __forceinline__ v16us ld_frag(const unsigned short* p0,
                                         const unsigned short* p1) {
  union { uint4 q[2]; v16us v; } f;
  f.q[0] = *(const uint4*)p0;
  f.q[1] = *(const uint4*)p1;
  return f.v;
}

__device__ __forceinline__ v8f zero8() {
  v8f z = {0.f,0.f,0.f,0.f,0.f,0.f,0.f,0.f};
  return z;
}

// ---- Tensor Data Mover: 2D bf16 tile (rows x 64) -> padded LDS ------------
// LDS row stride 72 elems = 144B = 32 dwords data + 4 dwords pad.
// This toolchain exposes the 6-arg builtin (clang-23 / therock-10.0 form):
//   (u32x4 g0, i32x8 g1, i32x4 g2, i32x4 g3, i32x8 pad, i32 cpol)
#if __has_builtin(__builtin_amdgcn_tensor_load_to_lds)
#define HAVE_TDM 1
__device__ __forceinline__ void tdm_load_k(const unsigned short* gsrc,
                                           unsigned lds_byte_off, int rows) {
  unsigned long long ga = (unsigned long long)(uintptr_t)gsrc;
  u32x4 g0;
  g0[0] = 1u;                                   // count=1 (valid user D#)
  g0[1] = lds_byte_off;                         // lds_addr
  g0[2] = (unsigned)(ga & 0xFFFFFFFFu);         // global_addr[31:0]
  g0[3] = (unsigned)((ga >> 32) & 0x1FFFFFFu)   // global_addr[56:32]
        | (2u << 30);                           // type = 2 ("image")
  i32x8 g1;
  g1[0] = (1 << 16)        // data_size = 1 (2 bytes)
        | (1 << 20)        // pad_enable
        | (4 << 22)        // pad_interval: 32 dwords between pads
        | (3 << 25);       // pad_amount:   4 dwords of pad
  g1[1] = (DH << 16);      // tensor_dim0[15:0] = 64
  g1[2] = (SEQ << 16);     // tensor_dim0[31:16]=0 | tensor_dim1[15:0]=8192
  g1[3] = (DH << 16);      // tensor_dim1[31:16]=0 | tile_dim0 = 64
  g1[4] = rows;            // tile_dim1 = rows, tile_dim2 = 0
  g1[5] = DH;              // tensor_dim0_stride[31:0] = 64
  g1[6] = 0;
  g1[7] = 0;
  i32x4 gz4 = {0, 0, 0, 0};
  i32x8 gz8 = {0, 0, 0, 0, 0, 0, 0, 0};
  __builtin_amdgcn_tensor_load_to_lds(g0, g1, gz4, gz4, gz8, 0);
}
#else
#define HAVE_TDM 0
#endif

// ---------------------------------------------------------------------------
// Kernel A: qkv = x @ w^T, output split per-head as bf16: q/k/v (b,h,n,d)
// grid (24 n-tiles, 256 m-tiles), 256 threads = 8 waves, BM=128 BN=64 BK=32
// ---------------------------------------------------------------------------
__global__ __launch_bounds__(256) void qkv_proj(const float* __restrict__ x,
                                                const float* __restrict__ w,
                                                unsigned short* __restrict__ qb,
                                                unsigned short* __restrict__ kb,
                                                unsigned short* __restrict__ vb) {
  extern __shared__ unsigned short smem[];
  unsigned short* XL = smem;             // 128 x (stride 48) [m][k] bf16
  unsigned short* WL = smem + 128 * 48;  // 64  x (stride 48) [n][k] bf16

  const int tid = threadIdx.x;
  const int lane = tid & 31;
  const int wave = tid >> 5;
  const int hl = lane >> 4;
  const int l16 = lane & 15;
  const int wm0 = (wave >> 1) * 32;
  const int wn0 = (wave & 1) * 32;
  const int m0 = blockIdx.y * 128;
  const int n0 = blockIdx.x * 64;

  v8f acc[2][2];
#pragma unroll
  for (int i = 0; i < 2; i++)
#pragma unroll
    for (int j = 0; j < 2; j++) acc[i][j] = zero8();

  for (int kk = 0; kk < DIMK; kk += 32) {
    // cooperative load X tile (128x32 fp32 -> bf16 LDS)
#pragma unroll
    for (int i = 0; i < 4; i++) {
      int f = tid + i * 256;              // 0..1023 float4 slots
      int r = f >> 3, c4 = f & 7;
      float4 xv = *(const float4*)(x + (size_t)(m0 + r) * DIMK + kk + c4 * 4);
      uint2 o = {cvt2(xv.x, xv.y), cvt2(xv.z, xv.w)};
      *(uint2*)(XL + r * 48 + c4 * 4) = o;
    }
    // cooperative load W tile (64x32 fp32 -> bf16 LDS, [n][k])
#pragma unroll
    for (int i = 0; i < 2; i++) {
      int f = tid + i * 256;              // 0..511
      int r = f >> 3, c4 = f & 7;
      float4 wv = *(const float4*)(w + (size_t)(n0 + r) * DIMK + kk + c4 * 4);
      uint2 o = {cvt2(wv.x, wv.y), cvt2(wv.z, wv.w)};
      *(uint2*)(WL + r * 48 + c4 * 4) = o;
    }
    if (kk + 32 < DIMK)
      __builtin_prefetch(x + (size_t)(m0 + (tid >> 1)) * DIMK + kk + 32 + (tid & 1) * 16, 0, 0);
    __syncthreads();

    v16us a[2], b[2];
#pragma unroll
    for (int mt = 0; mt < 2; mt++) {
      const unsigned short* pa = XL + (wm0 + mt * 16 + l16) * 48 + hl * 8;
      a[mt] = ld_frag(pa, pa + 16);
    }
#pragma unroll
    for (int nt = 0; nt < 2; nt++) {
      const unsigned short* pb = WL + (wn0 + nt * 16 + l16) * 48 + hl * 8;
      b[nt] = ld_frag(pb, pb + 16);
    }
#pragma unroll
    for (int mt = 0; mt < 2; mt++)
#pragma unroll
      for (int nt = 0; nt < 2; nt++)
        acc[mt][nt] = wmma_bf16(a[mt], b[nt], acc[mt][nt]);
    __syncthreads();
  }

  // epilogue: scatter into per-head q/k/v bf16 layouts
  const int which = blockIdx.x >> 3;      // 0:q 1:k 2:v (BN=64 == one head)
  const int hh = blockIdx.x & 7;
  unsigned short* dst = which == 0 ? qb : (which == 1 ? kb : vb);
  const int bidx = m0 >> 13;              // m0 / 8192
#pragma unroll
  for (int mt = 0; mt < 2; mt++)
#pragma unroll
    for (int nt = 0; nt < 2; nt++)
#pragma unroll
      for (int v = 0; v < 8; v++) {
        int Mg = m0 + wm0 + mt * 16 + v + hl * 8;
        int npos = Mg & (SEQ - 1);
        int d = wn0 + nt * 16 + l16;
        dst[(size_t)((bidx * HEADS + hh) * SEQ + npos) * DH + d] =
            f2bf(acc[mt][nt][v]);
      }
}

// ---------------------------------------------------------------------------
// Kernel B: kv = softmax_d( softmax_n(agent @ k^T) @ v )  -> bf16 (b,h,m,d)
// grid 32 (=b*h), 256 threads; each wave owns 32 agent rows; streaming online
// softmax over n in chunks of 128. K-chunks arrive via the Tensor Data Mover
// (double-buffered, overlapped with compute); V is loaded+transposed by VALU.
// ---------------------------------------------------------------------------
__global__ __launch_bounds__(256) void agent_kv(const float* __restrict__ agent,
                                                const unsigned short* __restrict__ kb,
                                                const unsigned short* __restrict__ vb,
                                                unsigned short* __restrict__ kvw) {
  extern __shared__ unsigned short smem[];
  unsigned short* KL0 = smem;                    // 128 x (stride 72) [n][d]
  unsigned short* KL1 = smem + 128 * 72;         // double buffer
  unsigned short* VT  = smem + 2 * 128 * 72;     // 64  x (stride 136) [d][n]
  unsigned short* PL  = VT + 64 * 136;           // 256 x (stride 136) [m][n]

  const int tid = threadIdx.x, lane = tid & 31, wave = tid >> 5;
  const int hl = lane >> 4, l16 = lane & 15;
  const int bh = blockIdx.x;
  const int hh = bh & 7;
  const size_t kvbase = (size_t)bh * SEQ * DH;
  const int wm0 = wave * 32;
#if HAVE_TDM
  const unsigned kl_off0 = (unsigned)(uintptr_t)(void*)KL0;
  const unsigned kl_off1 = (unsigned)(uintptr_t)(void*)KL1;
#endif

  // agent A-fragments, resident in registers for the whole kernel
  v16us afr[2][2];
#pragma unroll
  for (int mt = 0; mt < 2; mt++)
#pragma unroll
    for (int kf = 0; kf < 2; kf++) {
      v16us t;
      const float* ap =
          agent + (size_t)(hh * MAG + wm0 + mt * 16 + l16) * DH + kf * 32 + hl * 8;
#pragma unroll
      for (int e = 0; e < 8; e++) t[e] = f2bf(ap[e]);
#pragma unroll
      for (int e = 0; e < 8; e++) t[8 + e] = f2bf(ap[16 + e]);
      afr[mt][kf] = t;
    }

  v8f kvacc[2][4];
  float rmax[2][8], rsum[2][8];
#pragma unroll
  for (int mt = 0; mt < 2; mt++) {
#pragma unroll
    for (int dt = 0; dt < 4; dt++) kvacc[mt][dt] = zero8();
#pragma unroll
    for (int v = 0; v < 8; v++) { rmax[mt][v] = -3.0e38f; rsum[mt][v] = 0.f; }
  }

#if HAVE_TDM
  if (wave == 0) tdm_load_k(kb + kvbase, kl_off0, 128);   // chunk 0
#endif

  const int NCH = SEQ / 128;
  for (int ci = 0; ci < NCH; ci++) {
    const int nc = ci * 128;
#if HAVE_TDM
    // issue TDM for next K chunk into the other buffer (reads of that buffer
    // finished at the barrier that ended iteration ci-1)
    if (ci + 1 < NCH && wave == 0)
      tdm_load_k(kb + kvbase + (size_t)(nc + 128) * DH,
                 ((ci + 1) & 1) ? kl_off1 : kl_off0, 128);
#endif
    // cooperative load v chunk transposed [d][n] (overlaps the TDM)
#pragma unroll
    for (int i = 0; i < 4; i++) {
      int f = tid + i * 256;               // 0..1023 (128 rows x 8 groups)
      int r = f >> 3, g = f & 7;
      uint4 vq = *(const uint4*)(vb + kvbase + (size_t)(nc + r) * DH + g * 8);
      const unsigned short* vs = (const unsigned short*)&vq;
#pragma unroll
      for (int j = 0; j < 8; j++) VT[(g * 8 + j) * 136 + r] = vs[j];
    }
#if HAVE_TDM
    if (wave == 0) {
      if (ci + 1 < NCH) __builtin_amdgcn_s_wait_tensorcnt((short)1);
      else              __builtin_amdgcn_s_wait_tensorcnt((short)0);
    }
    const unsigned short* KL = (ci & 1) ? KL1 : KL0;
#else
    unsigned short* KL = KL0;
#pragma unroll
    for (int i = 0; i < 4; i++) {
      int f = tid + i * 256;
      int r = f >> 3, g = f & 7;
      uint4 kq = *(const uint4*)(kb + kvbase + (size_t)(nc + r) * DH + g * 8);
      *(uint4*)(KL + r * 72 + g * 8) = kq;
    }
#endif
    __syncthreads();

    // s = agent_tile @ k_chunk^T   (32 x 128 per wave)
    v8f s[2][8];
#pragma unroll
    for (int mt = 0; mt < 2; mt++)
#pragma unroll
      for (int nt = 0; nt < 8; nt++) {
        v8f a = zero8();
#pragma unroll
        for (int kf = 0; kf < 2; kf++) {
          const unsigned short* pb = KL + (nt * 16 + l16) * 72 + kf * 32 + hl * 8;
          a = wmma_bf16(afr[mt][kf], ld_frag(pb, pb + 16), a);
        }
        s[mt][nt] = a;
      }

    // online softmax update (rows are wave-private: 16-lane reductions only)
#pragma unroll
    for (int mt = 0; mt < 2; mt++)
#pragma unroll
      for (int v = 0; v < 8; v++) {
        float cm = -3.0e38f;
#pragma unroll
        for (int nt = 0; nt < 8; nt++) cm = fmaxf(cm, s[mt][nt][v]);
        cm = fmaxf(cm, __shfl_xor(cm, 1));
        cm = fmaxf(cm, __shfl_xor(cm, 2));
        cm = fmaxf(cm, __shfl_xor(cm, 4));
        cm = fmaxf(cm, __shfl_xor(cm, 8));
        float nm = fmaxf(rmax[mt][v], cm);
        float sc = __expf(rmax[mt][v] - nm);
        rmax[mt][v] = nm;
        float part = 0.f;
        int prow = (wm0 + mt * 16 + v + hl * 8) * 136;
#pragma unroll
        for (int nt = 0; nt < 8; nt++) {
          float e = __expf(s[mt][nt][v] - nm);
          part += e;
          PL[prow + nt * 16 + l16] = f2bf(e);
        }
        rsum[mt][v] = rsum[mt][v] * sc + part;
#pragma unroll
        for (int dt = 0; dt < 4; dt++) kvacc[mt][dt][v] *= sc;
      }

    // kvacc += P @ v_chunk   (wave-private P rows; same-wave LDS is in-order)
#pragma unroll
    for (int mt = 0; mt < 2; mt++)
#pragma unroll
      for (int kf = 0; kf < 4; kf++) {
        const unsigned short* pa = PL + (wm0 + mt * 16 + l16) * 136 + kf * 32 + hl * 8;
        v16us af = ld_frag(pa, pa + 16);
#pragma unroll
        for (int dt = 0; dt < 4; dt++) {
          const unsigned short* pb = VT + (dt * 16 + l16) * 136 + kf * 32 + hl * 8;
          kvacc[mt][dt] = wmma_bf16(af, ld_frag(pb, pb + 16), kvacc[mt][dt]);
        }
      }
    __syncthreads();
  }

  // finalize: normalize over n, then softmax over d, write bf16 kv
#pragma unroll
  for (int mt = 0; mt < 2; mt++)
#pragma unroll
    for (int v = 0; v < 8; v++) {
      float rs = rsum[mt][v];
      rs += __shfl_xor(rs, 1); rs += __shfl_xor(rs, 2);
      rs += __shfl_xor(rs, 4); rs += __shfl_xor(rs, 8);
      float inv = 1.0f / rs;
      float kvv[4];
      float dm = -3.0e38f;
#pragma unroll
      for (int dt = 0; dt < 4; dt++) {
        kvv[dt] = kvacc[mt][dt][v] * inv;
        dm = fmaxf(dm, kvv[dt]);
      }
      dm = fmaxf(dm, __shfl_xor(dm, 1));
      dm = fmaxf(dm, __shfl_xor(dm, 2));
      dm = fmaxf(dm, __shfl_xor(dm, 4));
      dm = fmaxf(dm, __shfl_xor(dm, 8));
      float e[4];
      float se = 0.f;
#pragma unroll
      for (int dt = 0; dt < 4; dt++) { e[dt] = __expf(kvv[dt] - dm); se += e[dt]; }
      se += __shfl_xor(se, 1); se += __shfl_xor(se, 2);
      se += __shfl_xor(se, 4); se += __shfl_xor(se, 8);
      float isr = 1.0f / se;
      int mg = wm0 + mt * 16 + v + hl * 8;
#pragma unroll
      for (int dt = 0; dt < 4; dt++)
        kvw[(size_t)bh * MAG * DH + (size_t)mg * DH + dt * 16 + l16] =
            f2bf(e[dt] * isr);
    }
}

// ---------------------------------------------------------------------------
// Kernel C: out = softmax_m(q @ agent^T * SCALE) @ kv   -> fp32 (b,n,h*d)
// grid (64 n-tiles, 32 bh), 256 threads; each wave handles 16 q rows.
// ---------------------------------------------------------------------------
__global__ __launch_bounds__(256) void out_kernel(const float* __restrict__ agent,
                                                  const unsigned short* __restrict__ qb,
                                                  const unsigned short* __restrict__ kvw,
                                                  float* __restrict__ out) {
  extern __shared__ unsigned short smem[];
  unsigned short* AG = smem;                // 256 x (stride 72)  [m][d]
  unsigned short* KT = smem + 256 * 72;     // 64  x (stride 264) [d][m]
  unsigned short* PL = KT + 64 * 264;       // 128 x (stride 264) [row][m]

  const int tid = threadIdx.x, lane = tid & 31, wave = tid >> 5;
  const int hl = lane >> 4, l16 = lane & 15;
  const int bh = blockIdx.y, hh = bh & 7, bidx = bh >> 3;
  const int rq = blockIdx.x * 128 + wave * 16;

  // agent -> LDS bf16 [m][d]
#pragma unroll
  for (int i = 0; i < 16; i++) {
    int f = tid + i * 256;                 // 0..4095 float4 slots
    int m = f >> 4, c4 = f & 15;
    float4 av = *(const float4*)(agent + (size_t)(hh * MAG + m) * DH + c4 * 4);
    uint2 o = {cvt2(av.x, av.y), cvt2(av.z, av.w)};
    *(uint2*)(AG + m * 72 + c4 * 4) = o;
  }
  // kv -> LDS transposed [d][m]
#pragma unroll
  for (int i = 0; i < 8; i++) {
    int f = tid + i * 256;                 // 0..2047
    int m = f >> 3, g = f & 7;
    uint4 kq = *(const uint4*)(kvw + (size_t)bh * MAG * DH + (size_t)m * DH + g * 8);
    const unsigned short* ks = (const unsigned short*)&kq;
#pragma unroll
    for (int j = 0; j < 8; j++) KT[(g * 8 + j) * 264 + m] = ks[j];
  }
  __syncthreads();

  // q A-fragments straight from global bf16
  v16us qf[2];
#pragma unroll
  for (int kf = 0; kf < 2; kf++) {
    const unsigned short* p =
        qb + ((size_t)bh * SEQ + rq + l16) * DH + kf * 32 + hl * 8;
    qf[kf] = ld_frag(p, p + 16);
  }

  // s = q @ agent^T  (16 x 256 per wave)
  v8f s[16];
#pragma unroll
  for (int nt = 0; nt < 16; nt++) {
    v8f a = zero8();
#pragma unroll
    for (int kf = 0; kf < 2; kf++) {
      const unsigned short* pb = AG + (nt * 16 + l16) * 72 + kf * 32 + hl * 8;
      a = wmma_bf16(qf[kf], ld_frag(pb, pb + 16), a);
    }
    s[nt] = a;
  }

  // softmax over m=256 (SCALE=1/8); keep unnormalized P, divide in epilogue
  float rsum[8];
#pragma unroll
  for (int v = 0; v < 8; v++) {
    float cm = -3.0e38f;
#pragma unroll
    for (int nt = 0; nt < 16; nt++) {
      s[nt][v] = s[nt][v] * 0.125f;
      cm = fmaxf(cm, s[nt][v]);
    }
    cm = fmaxf(cm, __shfl_xor(cm, 1));
    cm = fmaxf(cm, __shfl_xor(cm, 2));
    cm = fmaxf(cm, __shfl_xor(cm, 4));
    cm = fmaxf(cm, __shfl_xor(cm, 8));
    float ps = 0.f;
    int prow = (wave * 16 + v + hl * 8) * 264;
#pragma unroll
    for (int nt = 0; nt < 16; nt++) {
      float e = __expf(s[nt][v] - cm);
      ps += e;
      PL[prow + nt * 16 + l16] = f2bf(e);
    }
    ps += __shfl_xor(ps, 1); ps += __shfl_xor(ps, 2);
    ps += __shfl_xor(ps, 4); ps += __shfl_xor(ps, 8);
    rsum[v] = ps;
  }

  // out = P @ kv   (16 x 64 per wave)
  v8f oacc[4];
#pragma unroll
  for (int dt = 0; dt < 4; dt++) oacc[dt] = zero8();
#pragma unroll
  for (int kf = 0; kf < 8; kf++) {
    const unsigned short* pa = PL + (wave * 16 + l16) * 264 + kf * 32 + hl * 8;
    v16us af = ld_frag(pa, pa + 16);
#pragma unroll
    for (int dt = 0; dt < 4; dt++) {
      const unsigned short* pb = KT + (dt * 16 + l16) * 264 + kf * 32 + hl * 8;
      oacc[dt] = wmma_bf16(af, ld_frag(pb, pb + 16), oacc[dt]);
    }
  }

  // epilogue: normalize, write fp32 (b, n, h*d)
#pragma unroll
  for (int dt = 0; dt < 4; dt++)
#pragma unroll
    for (int v = 0; v < 8; v++) {
      int nrow = rq + v + hl * 8;
      float val = oacc[dt][v] / rsum[v];
      out[((size_t)bidx * SEQ + nrow) * (HEADS * DH) + hh * DH + dt * 16 + l16] = val;
    }
}

// ---------------------------------------------------------------------------
extern "C" void kernel_launch(void* const* d_in, const int* in_sizes, int n_in,
                              void* d_out, int out_size, void* d_ws, size_t ws_size,
                              hipStream_t stream) {
  const float* x = (const float*)d_in[0];
  const float* w = (const float*)d_in[1];
  const float* agent = (const float*)d_in[2];
  float* out = (float*)d_out;

  const size_t per = (size_t)BATCH * HEADS * SEQ * DH;   // 16,777,216 elems
  unsigned short* qb = (unsigned short*)d_ws;
  unsigned short* kb = qb + per;
  unsigned short* vb = kb + per;
  unsigned short* kvw = vb + per;                        // 524,288 elems

  size_t shA = (size_t)(128 * 48 + 64 * 48) * sizeof(unsigned short);
  hipLaunchKernelGGL(qkv_proj, dim3(24, 256), dim3(256), shA, stream,
                     x, w, qb, kb, vb);

  size_t shB = (size_t)(2 * 128 * 72 + 64 * 136 + 256 * 136) * sizeof(unsigned short);
  hipLaunchKernelGGL(agent_kv, dim3(32), dim3(256), shB, stream,
                     agent, kb, vb, kvw);

  size_t shC = (size_t)(256 * 72 + 64 * 264 + 128 * 264) * sizeof(unsigned short);
  hipLaunchKernelGGL(out_kernel, dim3(64, 32), dim3(256), shC, stream,
                     agent, qb, kvw, out);
}
